// GRU_30253749633385
// MI455X (gfx1250) — matrix operand order repaired and checked by reading, not compiled, not run on hardware
//
#include <hip/hip_runtime.h>
#include <hip/hip_bf16.h>
#include <stdint.h>

// Problem sizes from the reference (hardcoded there too).
#define BSZ 256
#define TSZ 512
#define HSZ 1024

typedef __attribute__((ext_vector_type(16))) __bf16          bf16x16;
typedef __attribute__((ext_vector_type(8)))  float           f32x8;
typedef __attribute__((ext_vector_type(8)))  unsigned short  u16x8;

__device__ __forceinline__ unsigned short f2bf(float f) {
    // round-to-nearest-even f32 -> bf16
    unsigned int u = __float_as_uint(f);
    u += 0x7fffu + ((u >> 16) & 1u);
    return (unsigned short)(u >> 16);
}

// Pack two f32 into a bf16 pair dword {hi:bf(b), lo:bf(a)} with one v_perm_b32.
__device__ __forceinline__ unsigned int pack_bf16(float a, float b) {
    const unsigned int ua = __float_as_uint(a) + 0x8000u;  // round-half-up
    const unsigned int ub = __float_as_uint(b) + 0x8000u;
    return __builtin_amdgcn_perm(ub, ua, 0x07060302u);     // {ub[3],ub[2],ua[3],ua[2]}
}

__device__ __forceinline__ bf16x16 loadB16(const unsigned short* p) {
    // two runs of 8 consecutive bf16 (p..p+7, p+16..p+23) per lane: two b128 loads
    union { bf16x16 v; u16x8 h[2]; } b;
    b.h[0] = *(const u16x8*)(p);
    b.h[1] = *(const u16x8*)(p + 16);
    return b.v;
}

// Load one K-chunk (32 wide): A fragment (16x32) + 4 B fragments (32x16).
// FIRST selects x/W half (f32 A source, convert) vs h/U half (bf16 A source, raw loads)
// at COMPILE TIME.
template <bool FIRST>
__device__ __forceinline__ void load_chunk(
    int kloc,
    const float* __restrict__ xrow, const unsigned short* __restrict__ hbrow,
    const unsigned short* __restrict__ zW, const unsigned short* __restrict__ zU,
    const unsigned short* __restrict__ hW, const unsigned short* __restrict__ hU,
    int laneHi,
    bf16x16& a, bf16x16& bz0, bf16x16& bz1, bf16x16& bh0, bf16x16& bh1)
{
    // ---- A fragment: lane holds two runs of 8 consecutive K values
    //      (K = c0..c0+7 and c0+16..c0+23).
    const int c0 = kloc + laneHi * 8;
    if (FIRST) {
        // x is f32 in memory: load 16 f32, convert/pack with v_perm.
        const float4 f0 = *(const float4*)(xrow + c0);
        const float4 f1 = *(const float4*)(xrow + c0 + 4);
        const float4 f2 = *(const float4*)(xrow + c0 + 16);
        const float4 f3 = *(const float4*)(xrow + c0 + 20);
        union { bf16x16 v; unsigned int d[8]; } af;
        af.d[0] = pack_bf16(f0.x, f0.y);
        af.d[1] = pack_bf16(f0.z, f0.w);
        af.d[2] = pack_bf16(f1.x, f1.y);
        af.d[3] = pack_bf16(f1.z, f1.w);
        af.d[4] = pack_bf16(f2.x, f2.y);
        af.d[5] = pack_bf16(f2.z, f2.w);
        af.d[6] = pack_bf16(f3.x, f3.y);
        af.d[7] = pack_bf16(f3.z, f3.w);
        a = af.v;
    } else {
        // h shadow is already bf16: two raw 16B loads, zero conversion ALU.
        union { bf16x16 v; u16x8 h[2]; } af;
        af.h[0] = *(const u16x8*)(hbrow + c0);
        af.h[1] = *(const u16x8*)(hbrow + c0 + 16);
        a = af.v;
    }

    // ---- B fragments: lane = column N, K = c0..c0+7 and c0+16..c0+23 (same as A).
    const unsigned short* wz = FIRST ? zW : zU;   // already offset to this wave's column base
    const unsigned short* wh = FIRST ? hW : hU;
    const int roff0 = kloc + laneHi * 8;
    const int roff1 = roff0 + 16 * HSZ;
    bz0 = loadB16(wz + roff0);
    bz1 = loadB16(wz + roff1);
    bh0 = loadB16(wh + roff0);
    bh1 = loadB16(wh + roff1);
}

// One-time: convert 4 weight matrices [H,H] f32 -> bf16, seed h0 (f32) and hb0 (bf16).
__global__ void gru_init(const float* __restrict__ Wz, const float* __restrict__ Uz,
                         const float* __restrict__ Wh, const float* __restrict__ Uh,
                         const float* __restrict__ ht,
                         unsigned short* __restrict__ w16,
                         float* __restrict__ h0, unsigned short* __restrict__ hb0) {
    const int HH = HSZ * HSZ;
    const int stride = gridDim.x * blockDim.x;
    for (int i = blockIdx.x * blockDim.x + threadIdx.x; i < 4 * HH; i += stride) {
        const int m = i / HH;
        const int e = i - m * HH;
        const float* src = (m == 0) ? Wz : (m == 1) ? Uz : (m == 2) ? Wh : Uh;
        w16[i] = f2bf(src[e]);
    }
    for (int i = blockIdx.x * blockDim.x + threadIdx.x; i < BSZ * HSZ; i += stride) {
        const float v = ht[i];
        h0[i]  = v;
        hb0[i] = f2bf(v);
    }
}

#define WMMA_SET(A_, BZ0_, BZ1_, BH0_, BH1_)                                                                 \
    accZ0 = __builtin_amdgcn_wmma_f32_16x16x32_bf16(false, A_, false, BZ0_, (short)0, accZ0, false, false);  \
    accZ1 = __builtin_amdgcn_wmma_f32_16x16x32_bf16(false, A_, false, BZ1_, (short)0, accZ1, false, false);  \
    accH0 = __builtin_amdgcn_wmma_f32_16x16x32_bf16(false, A_, false, BH0_, (short)0, accH0, false, false);  \
    accH1 = __builtin_amdgcn_wmma_f32_16x16x32_bf16(false, A_, false, BH1_, (short)0, accH1, false, false)

#define LOAD_CHUNK(FIRST_, KLOC_, A_, BZ0_, BZ1_, BH0_, BH1_)                                 \
    load_chunk<FIRST_>(KLOC_, xrow, hbrow, zWp, zUp, hWp, hUp, laneHi,                        \
                       A_, BZ0_, BZ1_, BH0_, BH1_)

// One GRU time step, fully fused:
//   gz = x_t.Wz^T + h.Uz^T ; gh = x_t.Wh^T + h.Uh^T ;
//   h_out = sigmoid(gz)*h + (1-sigmoid(gz))*tanh(gh)
// 64x64 tile per WG, 8 waves, each 16 rows x 32 cols x 2 gates.
// Two compile-time K phases (x-half f32, h-half bf16), register-pipelined across boundary.
__global__ __launch_bounds__(256)
void gru_step(const float* __restrict__ x, int t,
              const float* __restrict__ hin, const unsigned short* __restrict__ hbin,
              float* __restrict__ hout, unsigned short* __restrict__ hbout,
              const unsigned short* __restrict__ Wz16, const unsigned short* __restrict__ Uz16,
              const unsigned short* __restrict__ Wh16, const unsigned short* __restrict__ Uh16) {
    const int tid    = threadIdx.x;
    const int wave   = tid >> 5;
    const int lane   = tid & 31;
    const int laneN  = lane & 15;
    const int laneHi = lane >> 4;

    const int mBase = blockIdx.x * 64 + (wave & 3) * 16;   // global batch row base
    const int nBase = blockIdx.y * 64 + (wave >> 2) * 32;  // global hidden col base

    // Pull next step's x slice toward L2 while this step computes (scan critical path).
    // Only the blockIdx.y==0 column of WGs: 64 rows x 4KB, 8 cache lines per thread.
    if (blockIdx.y == 0 && t + 1 < TSZ) {
        const float* xn = x + ((size_t)(blockIdx.x * 64 + (tid >> 2)) * TSZ + (size_t)(t + 1)) * HSZ
                            + (size_t)(tid & 3) * 256;
        #pragma unroll
        for (int j = 0; j < 8; ++j)
            __builtin_prefetch(xn + j * 32, 0, 1);   // one 128B line each
    }

    f32x8 accZ0 = {}, accZ1 = {}, accH0 = {}, accH1 = {};

    const float*          xrow  = x    + ((size_t)(mBase + laneN) * TSZ + (size_t)t) * HSZ;
    const unsigned short* hbrow = hbin +  (size_t)(mBase + laneN) * HSZ;
    const size_t bcol = (size_t)(nBase + laneN) * HSZ;
    const unsigned short* zWp = Wz16 + bcol;
    const unsigned short* zUp = Uz16 + bcol;
    const unsigned short* hWp = Wh16 + bcol;
    const unsigned short* hUp = Uh16 + bcol;

    bf16x16 a0, bz00, bz10, bh00, bh10;   // pipeline slot 0
    bf16x16 a1, bz01, bz11, bh01, bh11;   // pipeline slot 1

    // ---- Phase A (x * W): chunks kloc = 0,32,...,992 (32 chunks)
    LOAD_CHUNK(true, 0, a0, bz00, bz10, bh00, bh10);
    for (int kk = 0; kk < HSZ - 64; kk += 64) {          // 15 iterations
        LOAD_CHUNK(true, kk + 32, a1, bz01, bz11, bh01, bh11);
        WMMA_SET(a0, bz00, bz10, bh00, bh10);
        LOAD_CHUNK(true, kk + 64, a0, bz00, bz10, bh00, bh10);
        WMMA_SET(a1, bz01, bz11, bh01, bh11);
    }
    // boundary peel: last x chunk, then first h-phase chunk
    LOAD_CHUNK(true, HSZ - 32, a1, bz01, bz11, bh01, bh11);
    WMMA_SET(a0, bz00, bz10, bh00, bh10);                 // chunk HSZ-64
    LOAD_CHUNK(false, 0, a0, bz00, bz10, bh00, bh10);     // phase B chunk 0
    WMMA_SET(a1, bz01, bz11, bh01, bh11);                 // chunk HSZ-32

    // ---- Phase B (h * U): chunks kloc = 0,32,...,992
    for (int kk = 0; kk < HSZ - 64; kk += 64) {
        LOAD_CHUNK(false, kk + 32, a1, bz01, bz11, bh01, bh11);
        WMMA_SET(a0, bz00, bz10, bh00, bh10);
        LOAD_CHUNK(false, kk + 64, a0, bz00, bz10, bh00, bh10);
        WMMA_SET(a1, bz01, bz11, bh01, bh11);
    }
    LOAD_CHUNK(false, HSZ - 32, a1, bz01, bz11, bh01, bh11);
    WMMA_SET(a0, bz00, bz10, bh00, bh10);
    WMMA_SET(a1, bz01, bz11, bh01, bh11);

    // ---- Epilogue: D layout = VGPR r: lanes 0-15 -> M=r, lanes 16-31 -> M=r+8, N=lane&15.
    // Writes both the f32 master h (for next step's blend) and the bf16 shadow
    // (for next step's phase-B A operand).
    #pragma unroll
    for (int r = 0; r < 8; ++r) {
        const int    row  = mBase + r + laneHi * 8;
        const size_t base = (size_t)row * HSZ + nBase + laneN;
        {
            const float hold = hin[base];
            const float z  = __builtin_amdgcn_rcpf(1.0f + __expf(-accZ0[r]));                       // sigmoid
            const float hc = 2.0f * __builtin_amdgcn_rcpf(1.0f + __expf(-2.0f * accH0[r])) - 1.0f;  // tanh
            const float hn = z * hold + (1.0f - z) * hc;
            hout[base]  = hn;
            hbout[base] = f2bf(hn);
        }
        {
            const float hold = hin[base + 16];
            const float z  = __builtin_amdgcn_rcpf(1.0f + __expf(-accZ1[r]));
            const float hc = 2.0f * __builtin_amdgcn_rcpf(1.0f + __expf(-2.0f * accH1[r])) - 1.0f;
            const float hn = z * hold + (1.0f - z) * hc;
            hout[base + 16]  = hn;
            hbout[base + 16] = f2bf(hn);
        }
    }
}

// Reference returns (h_final, h_final): write it twice.
__global__ void gru_out(const float* __restrict__ h, float* __restrict__ out) {
    const int i = blockIdx.x * blockDim.x + threadIdx.x;
    if (i < BSZ * HSZ) {
        out[i]             = h[i];
        out[BSZ * HSZ + i] = h[i];
    }
}

extern "C" void kernel_launch(void* const* d_in, const int* in_sizes, int n_in,
                              void* d_out, int out_size, void* d_ws, size_t ws_size,
                              hipStream_t stream) {
    // setup_inputs order: x, ht, Wz, Uz, Wr, Ur, Wh, Uh  (Wr/Ur unused, as in reference)
    const float* x  = (const float*)d_in[0];
    const float* ht = (const float*)d_in[1];
    const float* Wz = (const float*)d_in[2];
    const float* Uz = (const float*)d_in[3];
    const float* Wh = (const float*)d_in[6];
    const float* Uh = (const float*)d_in[7];

    // Workspace layout (11 MB total):
    //   [0, 8MB)     : 4x bf16 weight matrices (Wz,Uz,Wh,Uh), each H*H
    //   [8MB, 9MB)   : h buffer 0 (f32 [B,H])
    //   [9MB, 10MB)  : h buffer 1 (f32)
    //   [10MB,10.5M) : h bf16 shadow 0
    //   [10.5M,11M)  : h bf16 shadow 1
    char* ws = (char*)d_ws;
    unsigned short* w16  = (unsigned short*)ws;
    unsigned short* Wz16 = w16;
    unsigned short* Uz16 = w16 + (size_t)HSZ * HSZ;
    unsigned short* Wh16 = w16 + (size_t)2 * HSZ * HSZ;
    unsigned short* Uh16 = w16 + (size_t)3 * HSZ * HSZ;
    float*          h0   = (float*)(ws + (size_t)4 * HSZ * HSZ * sizeof(unsigned short));
    float*          h1   = h0 + (size_t)BSZ * HSZ;
    unsigned short* hb0  = (unsigned short*)(h1 + (size_t)BSZ * HSZ);
    unsigned short* hb1  = hb0 + (size_t)BSZ * HSZ;

    gru_init<<<1024, 256, 0, stream>>>(Wz, Uz, Wh, Uh, ht, w16, h0, hb0);

    const dim3 grid(BSZ / 64, HSZ / 64);  // 4 x 16 = 64 workgroups per step
    for (int t = 0; t < TSZ; ++t) {
        const float*          hin   = (t & 1) ? h1  : h0;
        const unsigned short* hbin  = (t & 1) ? hb1 : hb0;
        float*                hout  = (t & 1) ? h0  : h1;
        unsigned short*       hbout = (t & 1) ? hb0 : hb1;
        gru_step<<<grid, 256, 0, stream>>>(x, t, hin, hbin, hout, hbout,
                                           Wz16, Uz16, Wh16, Uh16);
    }
    // T=512 even: final state is in h0 (last step t=511 wrote h0).
    gru_out<<<(BSZ * HSZ + 255) / 256, 256, 0, stream>>>(h0, (float*)d_out);
}
